// SupConLossWithMemoryBank_46428596470022
// MI455X (gfx1250) — matrix operand order repaired and checked by reading, not compiled, not run on hardware
//
#include <hip/hip_runtime.h>

// ---------------------------------------------------------------------------
// SupCon loss with memory bank, CDNA5 (gfx1250, wave32) WMMA implementation.
//   N = 8704 rows (8192 bank + 512 batch), D = 128, TEMP = 0.07
//
// Math notes (exact, not approximations):
//  * The loss is invariant to the row-max stabilizer m (it cancels between the
//    pos term and log-sum-exp), and rows are unit vectors, so m = 1/T.
//  * sumpos_i = sum_{j same class, j != i} logit_ij
//             = (cf_i . classsum[label_i] - cf_i . cf_i) / T
//    -> computed by a tiny side kernel; removes pos-mask work and the
//       self-exclusion check from the O(N^2) epilogue.
//  * npos_i = hist[label_i] - 1 from a 100-bin class histogram.
//  * exp argument is bounded in [-41.3, 0] -> raw v_exp_f32 is safe (no
//    denormal-range fixup needed), so use __builtin_amdgcn_exp2f.
// Hot kernel epilogue per gram element: pk_fma(0.5) + exp + cmp + cndmask+add.
// ---------------------------------------------------------------------------

typedef __bf16 bf16_t;
typedef __attribute__((ext_vector_type(8)))  bf16_t bf16x8;
typedef __attribute__((ext_vector_type(16))) bf16_t v16bf;
typedef __attribute__((ext_vector_type(8)))  float  v8f;

#define N_TOT   8704
#define NBANK   8192
#define NBATCH  512
#define DDIM    128
#define NCLS    100
#define INV_T   14.285714285714286f            /* 1 / 0.07            */
#define KC      20.609929155556618f            /* (1/T) * log2(e)     */

// ---------------------------------------------------------------------------
// ws layout (bytes)
#define CFB_BYTES  ((size_t)2 * N_TOT * DDIM * sizeof(bf16_t))   /* 4,456,448 */
#define OFF_HIST   (CFB_BYTES)                                   /* 400 B     */
#define OFF_LACC   (OFF_HIST + 1024)                             /* 4 B       */
#define OFF_CSUM   (OFF_LACC + 1024)                             /* 102,400 B */
#define OFF_SUMP   (OFF_CSUM + (size_t)2 * NCLS * DDIM * 4)      /* 69,632 B  */
#define ZERO_FLTS  ((1024 + 1024 + 2 * NCLS * DDIM * 4) / 4)     /* 26,112    */

// ---------------------------------------------------------------------------
__global__ void supcon_zero_k(float* __restrict__ z) {
    int t = blockIdx.x * blockDim.x + threadIdx.x;
    if (t < ZERO_FLTS) z[t] = 0.0f;     // hist + lacc + csum
}

// Build bf16 cf matrices: branch0 = [bank2^T ; f1], branch1 = [bank1^T ; f2].
__global__ void supcon_convert_k(const float* __restrict__ features,
                                 const float* __restrict__ bank1,
                                 const float* __restrict__ bank2,
                                 bf16_t* __restrict__ cfb) {
    const int PER = N_TOT * (DDIM / 8);
    int t = blockIdx.x * blockDim.x + threadIdx.x;
    if (t >= 2 * PER) return;
    int branch = t / PER;
    int rem    = t % PER;
    int r      = rem % N_TOT;                        // row fastest -> coalesced
    int kc     = (rem / N_TOT) * 8;

    float vals[8];
    if (r < NBANK) {
        const float* bank = (branch == 0) ? bank2 : bank1;   // (D, BANK) row-major
        #pragma unroll
        for (int i = 0; i < 8; ++i)
            vals[i] = bank[(size_t)(kc + i) * NBANK + r];    // transpose read
    } else {
        int b = (r - NBANK) + branch * NBATCH;               // f1 or f2 row
        #pragma unroll
        for (int i = 0; i < 8; ++i)
            vals[i] = features[(size_t)b * DDIM + kc + i];
    }
    bf16x8 o;
    #pragma unroll
    for (int i = 0; i < 8; ++i) o[i] = (bf16_t)vals[i];
    *(bf16x8*)(cfb + (size_t)branch * N_TOT * DDIM + (size_t)r * DDIM + kc) = o;
}

// Class histogram over all 8704 labels.
__global__ void supcon_hist_k(const int* __restrict__ bank_labels,
                              const int* __restrict__ label,
                              int* __restrict__ hist) {
    int i = blockIdx.x * blockDim.x + threadIdx.x;
    if (i >= N_TOT) return;
    int lab = (i < NBANK) ? bank_labels[i] : label[i - NBANK];
    atomicAdd(&hist[lab], 1);
}

// Per-class feature sums: csum[branch][class][k] = sum of cf rows of class.
__launch_bounds__(256)
__global__ void supcon_csum_k(const bf16_t* __restrict__ cfb,
                              const int* __restrict__ bank_labels,
                              const int* __restrict__ label,
                              float* __restrict__ csum) {
    __shared__ float cs[NCLS * DDIM];                 // 51,200 B
    const int tid     = threadIdx.x;
    const int branch  = blockIdx.y;
    const int rowBase = blockIdx.x * 256;
    const bf16_t* cf  = cfb + (size_t)branch * N_TOT * DDIM;

    for (int i = tid; i < NCLS * DDIM; i += 256) cs[i] = 0.0f;
    __syncthreads();

    const int k   = tid & 127;
    const int sub = tid >> 7;                          // 0,1: two rows / iter
    for (int rr = sub; rr < 256; rr += 2) {
        int r   = rowBase + rr;
        int lab = (r < NBANK) ? bank_labels[r] : label[r - NBANK];
        float v = (float)cf[(size_t)r * DDIM + k];     // coalesced 256B reads
        atomicAdd(&cs[lab * DDIM + k], v);
    }
    __syncthreads();

    float* out = csum + (size_t)branch * NCLS * DDIM;
    for (int i = tid; i < NCLS * DDIM; i += 256) atomicAdd(&out[i], cs[i]);
}

// Per-row positive-logit sum: sumpos[b][i] = (cf_i.csum[lab_i] - cf_i.cf_i)/T.
__launch_bounds__(256)
__global__ void supcon_sumpos_k(const bf16_t* __restrict__ cfb,
                                const int* __restrict__ bank_labels,
                                const int* __restrict__ label,
                                const float* __restrict__ csum,
                                float* __restrict__ sumposW) {
    const int tid  = threadIdx.x;
    const int lane = tid & 31;
    const int wave = tid >> 5;
    const int gid  = blockIdx.x * 8 + wave;            // 0 .. 2*N_TOT-1
    const int branch = gid / N_TOT;
    const int row    = gid % N_TOT;
    const bf16_t* cf = cfb + (size_t)branch * N_TOT * DDIM;

    int lab = (row < NBANK) ? bank_labels[row] : label[row - NBANK];
    const float* csRow = csum + ((size_t)branch * NCLS + lab) * DDIM;

    float dcs = 0.0f, dself = 0.0f;
    #pragma unroll
    for (int i = 0; i < 4; ++i) {
        int k   = lane * 4 + i;
        float v = (float)cf[(size_t)row * DDIM + k];
        dcs   += v * csRow[k];
        dself += v * v;
    }
    #pragma unroll
    for (int off = 16; off >= 1; off >>= 1) {
        dcs   += __shfl_xor(dcs, off, 32);
        dself += __shfl_xor(dself, off, 32);
    }
    if (lane == 0) sumposW[gid] = (dcs - dself) * INV_T;
}

// ---------------------------------------------------------------------------
// Main gram kernel. Block = 256 threads (8 waves). Block owns 64 rows of one
// branch; loops over 68 column chunks of 128. Wave w: 2x2 register tile of
// 16x16 WMMA tiles (rows (w>>2)*32..+32, cols (w&3)*32..+32 within chunk):
// A and B fragments each reused twice -> 2.0 ds_load_b128 per WMMA, and the
// 4 WMMAs per K-step pipeline back-to-back through the XDL unit.
__launch_bounds__(256)
__global__ void supcon_gram_k(const bf16_t* __restrict__ cfb,
                              const int*   __restrict__ bank_labels,
                              const int*   __restrict__ label,
                              const int*   __restrict__ hist,
                              const float* __restrict__ sumposW,
                              float*       __restrict__ lacc) {
    __shared__ bf16_t colbuf[128][136];   // +8 halfs pad -> conflict-free
    __shared__ bf16_t rowbuf[64][136];
    __shared__ int    collab[128];
    __shared__ int    rowlabS[64];
    __shared__ float  snRow[64];

    const int tid     = threadIdx.x;
    const int branch  = blockIdx.y;
    const int rowBase = blockIdx.x * 64;
    const bf16_t* cf  = cfb + (size_t)branch * N_TOT * DDIM;

    // ---- stage the 64 block rows (once) -----------------------------------
    for (int idx = tid; idx < 64 * 16; idx += 256) {
        int r = idx >> 4, kc = (idx & 15) * 8;
        *(bf16x8*)(&rowbuf[r][kc]) =
            *(const bf16x8*)(cf + (size_t)(rowBase + r) * DDIM + kc);
    }
    if (tid < 64) {
        int r = rowBase + tid;
        rowlabS[tid] = (r < NBANK) ? bank_labels[r] : label[r - NBANK];
        snRow[tid]   = 0.0f;
    }
    __syncthreads();

    const int lane = tid & 31;
    const int wave = tid >> 5;
    const int rp   = wave >> 2;        // 0..1 : row-tile pair
    const int cp   = wave & 3;         // 0..3 : col-tile pair
    const int nlo  = lane & 15;
    const int hi   = lane >> 4;        // 0: lanes 0-15, 1: lanes 16-31

    int labr[16];
    #pragma unroll
    for (int i = 0; i < 2; ++i)
        #pragma unroll
        for (int g = 0; g < 8; ++g)
            labr[i * 8 + g] = rowlabS[(rp * 2 + i) * 16 + g + hi * 8];

    float sn[16];
    #pragma unroll
    for (int i = 0; i < 16; ++i) sn[i] = 0.0f;

    for (int cb = 0; cb < N_TOT / 128; ++cb) {
        const int colBase = cb * 128;
        __syncthreads();   // previous iteration finished reading colbuf
        for (int idx = tid; idx < 128 * 16; idx += 256) {
            int r = idx >> 4, kc = (idx & 15) * 8;
            *(bf16x8*)(&colbuf[r][kc]) =
                *(const bf16x8*)(cf + (size_t)(colBase + r) * DDIM + kc);
        }
        if (tid < 128) {
            int c = colBase + tid;
            collab[tid] = (c < NBANK) ? bank_labels[c] : label[c - NBANK];
        }
        __syncthreads();

        v8f acc[2][2];
        #pragma unroll
        for (int i = 0; i < 2; ++i)
            #pragma unroll
            for (int j = 0; j < 2; ++j)
                acc[i][j] = (v8f){0.f, 0.f, 0.f, 0.f, 0.f, 0.f, 0.f, 0.f};

        #pragma unroll
        for (int ks = 0; ks < 4; ++ks) {
            // A fragments (16x32 bf16), CDNA5 per-lane layout:
            //   lane M = lane&15 ; e0..7  -> K = ks*32 + hi*8 + e
            //                      e8..15 -> K = ks*32 + 16 + hi*8 + (e-8)
            v16bf afrag[2];
            #pragma unroll
            for (int i = 0; i < 2; ++i) {
                const bf16_t* ar = &rowbuf[(rp * 2 + i) * 16 + nlo][ks * 32 + hi * 8];
                bf16x8 a0 = *(const bf16x8*)(ar);
                bf16x8 a1 = *(const bf16x8*)(ar + 16);
                #pragma unroll
                for (int e = 0; e < 8; ++e) { afrag[i][e] = a0[e]; afrag[i][e + 8] = a1[e]; }
            }
            // B fragments (32x16 bf16): lane N = lane&15 ; e0..7 -> K = ks*32 + hi*8 + e, e8..15 -> K = ks*32 + 16 + hi*8 + (e-8)
            v16bf bfrag[2];
            #pragma unroll
            for (int j = 0; j < 2; ++j) {
                const bf16_t* br = &colbuf[(cp * 2 + j) * 16 + nlo][ks * 32 + hi * 8];
                bf16x8 b0 = *(const bf16x8*)(br);
                bf16x8 b1 = *(const bf16x8*)(br + 16);
                #pragma unroll
                for (int e = 0; e < 8; ++e) { bfrag[j][e] = b0[e]; bfrag[j][e + 8] = b1[e]; }
            }
            #pragma unroll
            for (int i = 0; i < 2; ++i)
                #pragma unroll
                for (int j = 0; j < 2; ++j)
                    acc[i][j] = __builtin_amdgcn_wmma_f32_16x16x32_bf16(
                        false, afrag[i], false, bfrag[j], (short)0, acc[i][j],
                        false, false);
        }

        // Epilogue (neg-sum only): exp(logit - 1/T) = exp2(acc*KC - KC).
        // Argument bounded in [-41.3, 0] -> raw v_exp_f32 is exact enough and
        // needs no denormal-range guard.
        // C/D layout: VGPR g -> M = g + 8*hi, N = lane&15.
        #pragma unroll
        for (int j = 0; j < 2; ++j) {
            const int labc = collab[(cp * 2 + j) * 16 + nlo];
            #pragma unroll
            for (int i = 0; i < 2; ++i) {
                #pragma unroll
                for (int g = 0; g < 8; ++g) {
                    float e = __builtin_amdgcn_exp2f(fmaf(acc[i][j][g], KC, -KC));
                    sn[i * 8 + g] += (labr[i * 8 + g] == labc) ? 0.0f : e;
                }
            }
        }
    }

    // ---- cross-lane reduce (16 lanes per half share a row) ----------------
    #pragma unroll
    for (int i = 0; i < 2; ++i) {
        #pragma unroll
        for (int g = 0; g < 8; ++g) {
            float s = sn[i * 8 + g];
            #pragma unroll
            for (int off = 8; off >= 1; off >>= 1) s += __shfl_xor(s, off, 32);
            if (nlo == 0)
                atomicAdd(&snRow[(rp * 2 + i) * 16 + g + hi * 8], s);
        }
    }
    __syncthreads();

    // ---- per-row loss, accumulate into global scalar ----------------------
    if (tid < 64) {
        int   row    = rowBase + tid;
        float npos   = (float)(hist[rowlabS[tid]] - 1);
        float sumpos = sumposW[branch * N_TOT + row];
        float mlpp   = sumpos / npos - INV_T - __logf(snRow[tid]);
        float contrib = -mlpp * (1.0f / (2.0f * (float)N_TOT));
        atomicAdd(lacc, contrib);
    }
}

__global__ void supcon_fin_k(const float* __restrict__ lacc, float* __restrict__ out) {
    if (threadIdx.x == 0) {
        float v = *lacc;
        out[0] = v; out[1] = v; out[2] = v;
    }
}

// ---------------------------------------------------------------------------
extern "C" void kernel_launch(void* const* d_in, const int* in_sizes, int n_in,
                              void* d_out, int out_size, void* d_ws, size_t ws_size,
                              hipStream_t stream) {
    const float* features    = (const float*)d_in[0];   // (1024, 1, 128)
    const int*   label       = (const int*)  d_in[1];   // (512,)
    const float* bank1       = (const float*)d_in[2];   // (128, 8192)
    const float* bank2       = (const float*)d_in[3];   // (128, 8192)
    const int*   bank_labels = (const int*)  d_in[4];   // (8192,)
    float*       out         = (float*)d_out;           // 3 floats

    char*   ws      = (char*)d_ws;
    bf16_t* cfb     = (bf16_t*)ws;                      // 2 x 8704 x 128 bf16
    int*    hist    = (int*)  (ws + OFF_HIST);          // 100 ints
    float*  lacc    = (float*)(ws + OFF_LACC);          // 1 float
    float*  csum    = (float*)(ws + OFF_CSUM);          // 2 x 100 x 128 f32
    float*  sumposW = (float*)(ws + OFF_SUMP);          // 2 x 8704 f32

    // 1) zero hist + lacc + csum (26,112 floats)
    supcon_zero_k<<<ZERO_FLTS / 256, 256, 0, stream>>>((float*)(ws + OFF_HIST));

    // 2) fp32 -> bf16 cf matrices (2 * 8704 * 16 = 278,528 threads)
    {
        int total  = 2 * N_TOT * (DDIM / 8);
        supcon_convert_k<<<(total + 255) / 256, 256, 0, stream>>>(features, bank1, bank2, cfb);
    }

    // 3) class histogram
    supcon_hist_k<<<(N_TOT + 255) / 256, 256, 0, stream>>>(bank_labels, label, hist);

    // 4) per-class feature sums (34 row-chunks x 2 branches)
    {
        dim3 grid(N_TOT / 256, 2);
        supcon_csum_k<<<grid, 256, 0, stream>>>(cfb, bank_labels, label, csum);
    }

    // 5) per-row positive-logit sums (one wave per row, 2*8704 rows)
    supcon_sumpos_k<<<(2 * N_TOT) / 8, 256, 0, stream>>>(cfb, bank_labels, label, csum, sumposW);

    // 6) WMMA gram + streaming neg log-sum-exp (136 row-blocks x 2 branches)
    {
        dim3 grid(N_TOT / 64, 2);
        supcon_gram_k<<<grid, 256, 0, stream>>>(cfb, bank_labels, label, hist, sumposW, lacc);
    }

    // 7) broadcast scalar loss to the 3 outputs
    supcon_fin_k<<<1, 32, 0, stream>>>(lacc, out);
}
